// Linear3Bit_65446711656922
// MI455X (gfx1250) — matrix operand order repaired and checked by reading, not compiled, not run on hardware
//
#include <hip/hip_runtime.h>
#include <hip/hip_bf16.h>
#include <stdint.h>

// GEMM: out[M,N] = x[M,K] * W^T (W: [N,K], 3-bit block-quantized, group=32) + bias
// M = 4096, N = 4096, K = 4096. Compute-bound (AI ~ 1000 FLOP/B): f16 WMMA
// 16x16x32, f32 accumulation, LDS double-buffered single-barrier pipeline.
// TILE_M=256 amortizes the expensive 3-bit dequant (shared across M-blocks).

typedef __attribute__((ext_vector_type(16))) _Float16 v16h;
typedef __attribute__((ext_vector_type(8)))  _Float16 v8h;
typedef __attribute__((ext_vector_type(8)))  float    v8f;

#define M_TOTAL 4096
#define N_TOTAL 4096
#define K_TOTAL 4096
#define TILE_M  256
#define TILE_N  128
#define TILE_K  64          // two quant groups per iteration
#define LDSTR   72          // padded stride in halves (144 B, 16B-aligned chunks)

struct AStage { float4 v[16]; };                      // 256x64 f32 tile / 256 thr
struct BStage { uint32_t d0, d1, d2; float norm; };   // one 3-bit group / thread

__device__ __forceinline__ void load_a_global(const float* __restrict__ x,
                                              int m0, int k0, int tid, AStage& s) {
#pragma unroll
  for (int i = 0; i < 16; ++i) {
    int idx = tid + i * 256;          // 0..4095
    int row = idx >> 4;               // 0..255
    int c4  = idx & 15;               // float4 column 0..15
    s.v[i] = *((const float4*)(x + (size_t)(m0 + row) * K_TOTAL + k0) + c4);
  }
}

__device__ __forceinline__ void store_a_lds(_Float16* __restrict__ As,
                                            int tid, const AStage& s) {
#pragma unroll
  for (int i = 0; i < 16; ++i) {
    int idx = tid + i * 256;
    int row = idx >> 4;
    int c4  = idx & 15;
    _Float16* dst = &As[row * LDSTR + c4 * 4];
    dst[0] = (_Float16)s.v[i].x; dst[1] = (_Float16)s.v[i].y;
    dst[2] = (_Float16)s.v[i].z; dst[3] = (_Float16)s.v[i].w;
  }
}

__device__ __forceinline__ void load_b_global(const uint8_t* __restrict__ wq,
                                              const float* __restrict__ wnorm,
                                              int n0, int k0, int tid, BStage& s) {
  int n = tid & 127;                  // output row within tile
  int g = tid >> 7;                   // which of the two K-groups
  size_t gidx = (size_t)(n0 + n) * (K_TOTAL / 32) + (k0 >> 5) + g;
  const uint32_t* p = (const uint32_t*)(wq + gidx * 12u);  // 12B record, 4B aligned
  s.d0 = p[0]; s.d1 = p[1]; s.d2 = p[2];
  s.norm = wnorm[gidx];
}

__device__ __forceinline__ void store_b_lds(_Float16* __restrict__ Bs,
                                            int tid, const BStage& s) {
  int n = tid & 127;
  int g = tid >> 7;
  uint64_t lo = ((uint64_t)s.d1 << 32) | s.d0;   // bits  0..63
  uint64_t hi = ((uint64_t)s.d2 << 32) | s.d1;   // bits 32..95
  float s2 = s.norm * (2.0f / 7.0f);             // w = q*(2n/7) - n
  float nn = s.norm;
  _Float16* dst = &Bs[n * LDSTR + g * 32];
#pragma unroll
  for (int j = 0; j < 32; ++j) {
    int bitp = 3 * j;
    uint32_t q = (bitp <= 61) ? ((uint32_t)(lo >> bitp) & 7u)
                              : ((uint32_t)(hi >> (bitp - 32)) & 7u);
    dst[j] = (_Float16)__builtin_fmaf((float)q, s2, -nn);
  }
}

__global__ __launch_bounds__(256)
void Linear3Bit_wmma_kernel(const float* __restrict__ x,
                            const uint8_t* __restrict__ wq,
                            const float* __restrict__ wnorm,
                            const float* __restrict__ bias,
                            float* __restrict__ out) {
  __shared__ _Float16 As[2][TILE_M * LDSTR];   // 73,728 B
  __shared__ _Float16 Bs[2][TILE_N * LDSTR];   // 36,864 B

  const int tid  = threadIdx.x;
  const int lane = tid & 31;
  const int wv   = tid >> 5;      // wave 0..7
  const int wm   = wv & 3;        // M-subtile offset = wm*64
  const int wn   = wv >> 2;       // N-subtile offset = wn*64
  const int lr   = lane & 15;
  const int lh   = lane >> 4;     // lane half

  const int m0 = blockIdx.y * TILE_M;
  const int n0 = blockIdx.x * TILE_N;

  v8f zero = {};
  v8f acc[4][4];
#pragma unroll
  for (int i = 0; i < 4; ++i)
#pragma unroll
    for (int j = 0; j < 4; ++j) acc[i][j] = zero;

  AStage a_stage;
  BStage b_stage;

  // Prologue: stage tile 0 into buffer 0.
  load_a_global(x, m0, 0, tid, a_stage);
  load_b_global(wq, wnorm, n0, 0, tid, b_stage);
  store_a_lds(As[0], tid, a_stage);
  store_b_lds(Bs[0], tid, b_stage);
  __syncthreads();

  int buf = 0;
  for (int k0 = 0; k0 < K_TOTAL; k0 += TILE_K) {
    const bool has_next = (k0 + TILE_K) < K_TOTAL;

    // (a) Issue next tile's global loads early (latency hidden behind WMMAs).
    if (has_next) {
      load_a_global(x, m0, k0 + TILE_K, tid, a_stage);
      load_b_global(wq, wnorm, n0, k0 + TILE_K, tid, b_stage);
    }

    // (b) Compute on current buffer: 2 K-substeps x 16 WMMAs.
    const _Float16* Ab = As[buf];
    const _Float16* Bb = Bs[buf];
#pragma unroll
    for (int ks = 0; ks < 2; ++ks) {
      const int kof = ks * 32;

      // Load all 4 B fragments once; stream A fragments to cap liveness.
      v16h bfrag[4];
#pragma unroll
      for (int j = 0; j < 4; ++j) {
        // B fragment, contiguous k order (matches A): lane half 0: K 0-15 of column; half 1: K 16-31.
        int bcol = wn * 64 + j * 16 + lr;
        union { v16h v; v8h h[2]; } u;
        u.h[0] = *(const v8h*)&Bb[bcol * LDSTR + kof + lh * 16];
        u.h[1] = *(const v8h*)&Bb[bcol * LDSTR + kof + lh * 16 + 8];
        bfrag[j] = u.v;
      }
#pragma unroll
      for (int i = 0; i < 4; ++i) {
        // A fragment, contiguous k order (matches B): lanes 0-15: K 0-15; lanes 16-31: K 16-31.
        int arow = wm * 64 + i * 16 + lr;
        union { v16h v; v8h h[2]; } u;
        u.h[0] = *(const v8h*)&Ab[arow * LDSTR + kof + lh * 16];
        u.h[1] = *(const v8h*)&Ab[arow * LDSTR + kof + lh * 16 + 8];
        v16h afrag = u.v;
#pragma unroll
        for (int j = 0; j < 4; ++j)
          acc[i][j] = __builtin_amdgcn_wmma_f32_16x16x32_f16(
              false, afrag, false, bfrag[j], (short)0, acc[i][j],
              false, false);
      }
    }

    // (c) Convert/unpack staged data into the other buffer.
    if (has_next) {
      store_a_lds(As[buf ^ 1], tid, a_stage);
      store_b_lds(Bs[buf ^ 1], tid, b_stage);
      // (d) Single barrier: publishes buf^1 writes AND retires this
      // iteration's buf reads before next iteration overwrites buf.
      __syncthreads();
      buf ^= 1;
    }
  }

  // Epilogue: bias + store (C/D layout: VGPR r -> row r, +8 for lane half 1).
  float bv[4];
#pragma unroll
  for (int j = 0; j < 4; ++j) bv[j] = bias[n0 + wn * 64 + j * 16 + lr];

#pragma unroll
  for (int i = 0; i < 4; ++i) {
#pragma unroll
    for (int j = 0; j < 4; ++j) {
      int col     = n0 + wn * 64 + j * 16 + lr;
      int rowbase = m0 + wm * 64 + i * 16 + lh * 8;
#pragma unroll
      for (int r = 0; r < 8; ++r) {
        out[(size_t)(rowbase + r) * N_TOTAL + col] = acc[i][j][r] + bv[j];
      }
    }
  }
}

extern "C" void kernel_launch(void* const* d_in, const int* in_sizes, int n_in,
                              void* d_out, int out_size, void* d_ws, size_t ws_size,
                              hipStream_t stream) {
  const float*   xp    = (const float*)d_in[0];
  const uint8_t* wqp   = (const uint8_t*)d_in[1];
  const float*   wnp   = (const float*)d_in[2];
  const float*   biasp = (const float*)d_in[3];
  float*         outp  = (float*)d_out;

  dim3 grid(N_TOTAL / TILE_N, M_TOTAL / TILE_M);  // 32 x 16
  Linear3Bit_wmma_kernel<<<grid, 256, 0, stream>>>(xp, wqp, wnp, biasp, outp);
}
